// MultiLocAwareAttention_32727650795938
// MI455X (gfx1250) — matrix operand order repaired and checked by reading, not compiled, not run on hardware
//
#include <hip/hip_runtime.h>

// Problem constants (match reference)
#define BB 4
#define SS 2048
#define DD 512
#define HH 8
#define KK 10
#define DH 64

typedef __attribute__((ext_vector_type(16))) __bf16 v16bf;
typedef __attribute__((ext_vector_type(8)))  __bf16 v8bf;
typedef __attribute__((ext_vector_type(8)))  float  v8f;

#define WMMA_BF16(a, b, c) \
  __builtin_amdgcn_wmma_f32_16x16x32_bf16(false, (a), false, (b), (short)0, (c), false, false)

// ---------------------------------------------------------------------------
// TDM support (ROCm7.2 clang-22: 5-arg builtin; upstream clang-23: 6-arg)
// ---------------------------------------------------------------------------
#if defined(__has_builtin)
#if __has_builtin(__builtin_amdgcn_tensor_load_to_lds) && \
    __has_builtin(__builtin_amdgcn_s_wait_tensorcnt)
#define HAVE_TDM 1
#endif
#endif
#ifndef HAVE_TDM
#define HAVE_TDM 0
#endif

#if HAVE_TDM
typedef __attribute__((ext_vector_type(4))) unsigned int tdm_u32x4;
typedef __attribute__((ext_vector_type(8))) int          tdm_i32x8;
typedef __attribute__((ext_vector_type(4))) int          tdm_i32x4;

// DMA a 2D tile (tile_d0 contiguous elems x tile_d1 rows) of a row-major
// [d1 x d0] tensor into LDS at lds_off. D# per CDNA5 ISA ch.8 (groups 2/3 zero).
// dsz_code: 0=1B, 1=2B, 2=4B, 3=8B.
__device__ __forceinline__ void tdm_load_2d(const void* gsrc, unsigned lds_off,
                                            unsigned d0, unsigned d1,
                                            unsigned tile_d0, unsigned tile_d1,
                                            unsigned long long stride0,
                                            unsigned dsz_code) {
  unsigned long long ga = (unsigned long long)(uintptr_t)gsrc;
  tdm_u32x4 g0;
  g0[0] = 1u;                                          // count=1, user descriptor
  g0[1] = lds_off;                                     // lds_addr (bytes)
  g0[2] = (unsigned)(ga & 0xffffffffu);                // global_addr[31:0]
  g0[3] = (unsigned)((ga >> 32) & 0x01ffffffu)         // global_addr[56:32]
        | (2u << 30);                                  // type=2 ("image")
  tdm_i32x8 g1;
  g1[0] = (int)(dsz_code << 16);                       // workgroup_mask=0, data_size
  g1[1] = (int)((d0 & 0xffffu) << 16);                 // tensor_dim0[15:0]
  g1[2] = (int)(((d0 >> 16) & 0xffffu)                 // tensor_dim0[31:16]
        | ((d1 & 0xffffu) << 16));                     // tensor_dim1[15:0]
  g1[3] = (int)(((d1 >> 16) & 0xffffu)                 // tensor_dim1[31:16]
        | ((tile_d0 & 0xffffu) << 16));                // tile_dim0
  g1[4] = (int)(tile_d1 & 0xffffu);                    // tile_dim1; tile_dim2=0
  g1[5] = (int)(unsigned)(stride0 & 0xffffffffu);      // tensor_dim0_stride[31:0]
  g1[6] = (int)(unsigned)((stride0 >> 32) & 0xffffu);  // stride[47:32]; dim1_stride=0
  g1[7] = 0;
  tdm_i32x4 z4 = {0, 0, 0, 0};
#if defined(__clang_major__) && (__clang_major__ >= 23)
  tdm_i32x8 z8 = {0, 0, 0, 0, 0, 0, 0, 0};
  __builtin_amdgcn_tensor_load_to_lds(g0, g1, z4, z4, z8, 0);
#else
  __builtin_amdgcn_tensor_load_to_lds(g0, g1, z4, z4, 0);
#endif
}
#endif  // HAVE_TDM

// Fallback: cooperative stage of a 64(row) x 32(k) bf16 tile of WT[n][k] into LDS
__device__ __forceinline__ void stage_tile_bf16_coop(const __bf16* __restrict__ WT, int ldk,
                                                     int kc, int n0, __bf16* tile) {
  const int t = threadIdx.x;
  unsigned* dst = (unsigned*)tile;
#pragma unroll
  for (int i = 0; i < 4; ++i) {
    int idx = i * 256 + t;                 // 1024 dwords = 2048 halves
    int row = idx >> 4;                    // 16 dwords per 32-half row
    int cw  = idx & 15;
    dst[idx] = ((const unsigned*)(WT + (size_t)(n0 + row) * ldk + kc))[cw];
  }
}

// ---------------------------------------------------------------------------
// Fragment builders (v_wmma_f32_16x16x32_bf16 operands; contiguous k order used by A and B here)
// A 16x32 (MxK): lane holds row M=lane&15; elem e -> K=e+(lane&16?16:0)
// B 32x16 (KxN): lane holds col N=lane&15; elem e -> K=e+(lane&16?16:0)
// ---------------------------------------------------------------------------
__device__ __forceinline__ v16bf a_frag_f32(const float* __restrict__ row, int kc, int lane) {
  const int hi8 = (lane & 16) ? 16 : 0;
  const float4* r0 = (const float4*)(row + kc + hi8);
  const float4* r1 = (const float4*)(row + kc + 8 + hi8);
  float4 a0 = r0[0], a1 = r0[1], b0 = r1[0], b1 = r1[1];
  v16bf f;
  f[0] = (__bf16)a0.x; f[1] = (__bf16)a0.y; f[2] = (__bf16)a0.z; f[3] = (__bf16)a0.w;
  f[4] = (__bf16)a1.x; f[5] = (__bf16)a1.y; f[6] = (__bf16)a1.z; f[7] = (__bf16)a1.w;
  f[8] = (__bf16)b0.x; f[9] = (__bf16)b0.y; f[10] = (__bf16)b0.z; f[11] = (__bf16)b0.w;
  f[12] = (__bf16)b1.x; f[13] = (__bf16)b1.y; f[14] = (__bf16)b1.z; f[15] = (__bf16)b1.w;
  return f;
}

__device__ __forceinline__ v16bf a_frag_bf16(const __bf16* __restrict__ row, int kc, int lane) {
  const int hi8 = (lane & 16) ? 16 : 0;
  v8bf lo = *(const v8bf*)(row + kc + hi8);
  v8bf hi = *(const v8bf*)(row + kc + 8 + hi8);
  return __builtin_shufflevector(lo, hi, 0, 1, 2, 3, 4, 5, 6, 7,
                                 8, 9, 10, 11, 12, 13, 14, 15);
}

// B fragment where memory holds B^T rows contiguously (rowptr = row n, contiguous k)
__device__ __forceinline__ v16bf b_frag_bf16_row(const __bf16* __restrict__ rowptr,
                                                 int kc, int lane) {
  const int hi16 = (lane & 16) ? 16 : 0;
  return *(const v16bf*)(rowptr + kc + hi16);
}

// B fragment from LDS bf16 tile laid out transposed: [n(64)][k(32)]
__device__ __forceinline__ v16bf b_frag_lds_bf16(const __bf16* tile, int nlocal, int lane) {
  const int hi16 = (lane & 16) ? 16 : 0;
  return *(const v16bf*)(tile + nlocal * 32 + hi16);   // 32B contiguous -> ds_load_b128 x2
}

// ---------------- Prep: f32 -> bf16 copy ---------------------------------------------
__global__ void cvt_bf16_kernel(const float* __restrict__ src, __bf16* __restrict__ dst) {
  int idx = blockIdx.x * 256 + threadIdx.x;
  dst[idx] = (__bf16)src[idx];
}

// ---------------- Prep: W[k][n] (n=512) -> WT[n][k] bf16 ------------------------------
__global__ void transpose_bf16_kernel(const float* __restrict__ W, __bf16* __restrict__ WT,
                                      int Ktot) {
  int idx = blockIdx.x * 256 + threadIdx.x;   // over Ktot*512, coalesced on read
  int k = idx >> 9;
  int n = idx & 511;
  WT[(size_t)n * Ktot + k] = (__bf16)W[idx];
}

// ---------------- K1: conv1d(prev_align) @ wu_w + wu_b -> loc (B,S,D head-major) ------
__global__ void loc_kernel(const float* __restrict__ pa, const float* __restrict__ cw,
                           const float* __restrict__ cb, const float* __restrict__ wu_w,
                           const float* __restrict__ wu_b, float* __restrict__ loc) {
  int tid = blockIdx.x * blockDim.x + threadIdx.x;  // B*H*S*DH = 2^22 threads
  int d  = tid & (DH - 1);
  int s  = (tid >> 6) & (SS - 1);
  int hb = tid >> 17;                               // b*H + h
  int h = hb & (HH - 1);
  int b = hb >> 3;
  const float* row = pa + (size_t)hb * SS;
  float left  = (s > 0)      ? row[s - 1] : 0.f;
  float mid   = row[s];
  float right = (s < SS - 1) ? row[s + 1] : 0.f;
  float acc = wu_b[d];
#pragma unroll
  for (int c = 0; c < KK; ++c) {
    float conv = cw[c * 3 + 0] * left + cw[c * 3 + 1] * mid + cw[c * 3 + 2] * right + cb[c];
    acc += conv * wu_w[c * DH + d];
  }
  loc[((size_t)(b * SS + s)) * DD + h * DH + d] = acc;
}

// ---------------- K2: projections -> qb (bf16), vb (bf16), vT (bf16 transposed) -------
// A = bf16 activations, B = bf16 transposed weights staged to LDS (TDM).
__global__ void proj_kernel(const __bf16* __restrict__ qact, const __bf16* __restrict__ vact,
                            const __bf16* __restrict__ wqT, const __bf16* __restrict__ wvT,
                            const float* __restrict__ bq, const float* __restrict__ bv,
                            const float* __restrict__ loc,
                            __bf16* __restrict__ qb, __bf16* __restrict__ vb,
                            __bf16* __restrict__ vT) {
  __shared__ __bf16 wtile[64 * 32];               // [n][k], 4KB
  const int lane = threadIdx.x & 31;
  const int wave = threadIdx.x >> 5;
  const bool isQ = (blockIdx.z == 0);
  const __bf16* A    = isQ ? qact : vact;
  const __bf16* WT   = isQ ? wqT : wvT;
  const float*  bias = isQ ? bq : bv;

  const int m0   = (blockIdx.x * 8 + wave) * 16;
  const int n0   = blockIdx.y * 64;
  const int mrow = m0 + (lane & 15);
  const int ncol = lane & 15;
  const __bf16* arow = A + (size_t)mrow * DD;

  v8f acc[4] = {};
  for (int kc = 0; kc < DD; kc += 32) {
    __syncthreads();                              // previous tile fully consumed
#if HAVE_TDM
    if (threadIdx.x < 32) {                       // one DMA per block (EXEC ignored)
      tdm_load_2d(WT + (size_t)n0 * DD + kc, (unsigned)(uintptr_t)wtile,
                  DD, DD, 32, 64, DD, 1u /*2B*/);
      __builtin_amdgcn_s_wait_tensorcnt(0);
    }
#else
    stage_tile_bf16_coop(WT, DD, kc, n0, wtile);
#endif
    __syncthreads();                              // tile visible to all waves
    v16bf a = a_frag_bf16(arow, kc, lane);
#pragma unroll
    for (int j = 0; j < 4; ++j) {
      v16bf bf = b_frag_lds_bf16(wtile, j * 16 + ncol, lane);
      acc[j] = WMMA_BF16(a, bf, acc[j]);
    }
  }
  const int mstore = m0 + ((lane & 16) ? 8 : 0);
#pragma unroll
  for (int j = 0; j < 4; ++j) {
    int n = n0 + j * 16 + ncol;
    float bn = bias[n];
#pragma unroll
    for (int r = 0; r < 8; ++r) {
      int m = mstore + r;
      float v = acc[j][r] + bn;
      if (isQ) {
        v += loc[(size_t)m * DD + n];
        qb[(size_t)m * DD + n] = (__bf16)v;
      } else {
        __bf16 h16 = (__bf16)v;
        vb[(size_t)m * DD + n] = h16;
        int h = n >> 6, d = n & 63;
        int b = m >> 11, s = m & (SS - 1);
        vT[(((size_t)b * HH + h) * DH + d) * SS + s] = h16;   // (b,h,d,s)
      }
    }
  }
}

// ---------------- K3: scores = qh @ vh^T / 8 -> align region of d_out -----------------
__global__ void scores_kernel(const __bf16* __restrict__ qb, const __bf16* __restrict__ vb,
                              float* __restrict__ alignOut) {
  const int lane = threadIdx.x & 31;
  const int wave = threadIdx.x >> 5;
  const int z = blockIdx.z;            // z = h*B + b (matches align.reshape(H*B,S,S))
  const int h = z >> 2;
  const int b = z & 3;
  const int m0   = (blockIdx.x * 8 + wave) * 16;
  const int n0   = blockIdx.y * 64;
  const int mrow = m0 + (lane & 15);
  const int ncol = lane & 15;
  const __bf16* arow = qb + ((size_t)(b * SS + mrow)) * DD + h * DH;

  v8f acc[4] = {};
#pragma unroll
  for (int kc = 0; kc < DH; kc += 32) {
    v16bf a = a_frag_bf16(arow, kc, lane);
#pragma unroll
    for (int j = 0; j < 4; ++j) {
      const __bf16* brow = vb + ((size_t)(b * SS + n0 + j * 16 + ncol)) * DD + h * DH;
      v16bf bf = b_frag_bf16_row(brow, kc, lane);    // vh^T: row n contiguous in k
      acc[j] = WMMA_BF16(a, bf, acc[j]);
    }
  }
  const float scale = 0.125f;                        // 1/sqrt(64)
  const int mstore = m0 + ((lane & 16) ? 8 : 0);
  float* base = alignOut + (size_t)z * SS * SS;
#pragma unroll
  for (int j = 0; j < 4; ++j) {
    int n = n0 + j * 16 + ncol;
#pragma unroll
    for (int r = 0; r < 8; ++r)
      base[(size_t)(mstore + r) * SS + n] = acc[j][r] * scale;
  }
}

// ---------------- K4: row softmax in-place on align (rows of 2048) --------------------
__global__ void softmax_kernel(float* __restrict__ al) {
  __shared__ float red[8];
  float* p = al + (size_t)blockIdx.x * SS;
  const int t = threadIdx.x;
  float v[8];
  float m = -3.402823466e38f;
#pragma unroll
  for (int i = 0; i < 8; ++i) { v[i] = p[t + i * 256]; m = fmaxf(m, v[i]); }
#pragma unroll
  for (int off = 16; off >= 1; off >>= 1) m = fmaxf(m, __shfl_xor(m, off, 32));
  if ((t & 31) == 0) red[t >> 5] = m;
  __syncthreads();
  m = red[0];
#pragma unroll
  for (int w = 1; w < 8; ++w) m = fmaxf(m, red[w]);
  float s = 0.f;
#pragma unroll
  for (int i = 0; i < 8; ++i) { v[i] = __expf(v[i] - m); s += v[i]; }
#pragma unroll
  for (int off = 16; off >= 1; off >>= 1) s += __shfl_xor(s, off, 32);
  __syncthreads();
  if ((t & 31) == 0) red[t >> 5] = s;
  __syncthreads();
  s = 0.f;
#pragma unroll
  for (int w = 0; w < 8; ++w) s += red[w];
  float inv = 1.f / s;
#pragma unroll
  for (int i = 0; i < 8; ++i) p[t + i * 256] = v[i] * inv;
}

// ---------------- K5: ctx = align @ vh -> (B,S,D head-major, bf16) --------------------
__global__ void ctx_kernel(const float* __restrict__ al, const __bf16* __restrict__ vT,
                           __bf16* __restrict__ ctxb) {
  const int lane = threadIdx.x & 31;
  const int wave = threadIdx.x >> 5;
  const int z = blockIdx.z;  const int h = z >> 2;  const int b = z & 3;
  const int m0   = (blockIdx.x * 8 + wave) * 16;
  const int mrow = m0 + (lane & 15);
  const int ncol = lane & 15;
  const float*  arow = al + ((size_t)(z * SS + mrow)) * SS;
  const __bf16* vTrow[4];
#pragma unroll
  for (int j = 0; j < 4; ++j)
    vTrow[j] = vT + (((size_t)b * HH + h) * DH + j * 16 + ncol) * SS;  // row d, contiguous k

  v8f acc[4] = {};
  for (int kc = 0; kc < SS; kc += 32) {
    __builtin_prefetch(arow + kc + 128, 0, 0);       // global_prefetch_b8 streaming hint
    v16bf a = a_frag_f32(arow, kc, lane);
#pragma unroll
    for (int j = 0; j < 4; ++j) {
      v16bf bf = b_frag_bf16_row(vTrow[j], kc, lane);
      acc[j] = WMMA_BF16(a, bf, acc[j]);
    }
  }
  const int mstore = m0 + ((lane & 16) ? 8 : 0);
#pragma unroll
  for (int j = 0; j < 4; ++j) {
    int n = j * 16 + ncol;
#pragma unroll
    for (int r = 0; r < 8; ++r)
      ctxb[((size_t)(b * SS + mstore + r)) * DD + h * DH + n] = (__bf16)acc[j][r];
  }
}

// ---------------- K6: out = [ctx, query] @ fc_w + fc_b (pre-LayerNorm) ----------------
__global__ void fc_kernel(const __bf16* __restrict__ ctxb, const __bf16* __restrict__ qact,
                          const __bf16* __restrict__ fcT, const float* __restrict__ fcb,
                          float* __restrict__ out) {
  __shared__ __bf16 wtile[64 * 32];
  const int lane = threadIdx.x & 31;
  const int wave = threadIdx.x >> 5;
  const int m0   = (blockIdx.x * 8 + wave) * 16;
  const int n0   = blockIdx.y * 64;
  const int mrow = m0 + (lane & 15);
  const int ncol = lane & 15;
  const __bf16* crow = ctxb + (size_t)mrow * DD;
  const __bf16* qrow = qact + (size_t)mrow * DD;

  v8f acc[4] = {};
  for (int kc = 0; kc < 2 * DD; kc += 32) {
    __syncthreads();
#if HAVE_TDM
    if (threadIdx.x < 32) {
      tdm_load_2d(fcT + (size_t)n0 * (2 * DD) + kc, (unsigned)(uintptr_t)wtile,
                  2 * DD, DD, 32, 64, 2 * DD, 1u /*2B*/);
      __builtin_amdgcn_s_wait_tensorcnt(0);
    }
#else
    stage_tile_bf16_coop(fcT, 2 * DD, kc, n0, wtile);
#endif
    __syncthreads();
    // concat([ctx, query]) along K: kc<512 reads ctx, else query at (kc-512)
    const __bf16* arow = (kc < DD) ? crow : (qrow - DD);
    v16bf a = a_frag_bf16(arow, kc, lane);
#pragma unroll
    for (int j = 0; j < 4; ++j) {
      v16bf bf = b_frag_lds_bf16(wtile, j * 16 + ncol, lane);
      acc[j] = WMMA_BF16(a, bf, acc[j]);
    }
  }
  const int mstore = m0 + ((lane & 16) ? 8 : 0);
#pragma unroll
  for (int j = 0; j < 4; ++j) {
    int n = n0 + j * 16 + ncol;
    float bn = fcb[n];
#pragma unroll
    for (int r = 0; r < 8; ++r)
      out[(size_t)(mstore + r) * DD + n] = acc[j][r] + bn;
  }
}

// ---------------- K7: LayerNorm in-place on out rows of 512 ---------------------------
__global__ void ln_kernel(float* __restrict__ out, const float* __restrict__ g,
                          const float* __restrict__ bt) {
  __shared__ float rs[8], rq[8];
  float* p = out + (size_t)blockIdx.x * DD;
  const int t = threadIdx.x;
  float x0 = p[t], x1 = p[t + 256];
  float s = x0 + x1, q = x0 * x0 + x1 * x1;
#pragma unroll
  for (int off = 16; off >= 1; off >>= 1) {
    s += __shfl_xor(s, off, 32);
    q += __shfl_xor(q, off, 32);
  }
  if ((t & 31) == 0) { rs[t >> 5] = s; rq[t >> 5] = q; }
  __syncthreads();
  s = 0.f; q = 0.f;
#pragma unroll
  for (int w = 0; w < 8; ++w) { s += rs[w]; q += rq[w]; }
  float mean = s * (1.f / DD);
  float var  = q * (1.f / DD) - mean * mean;
  float inv  = rsqrtf(var + 1e-5f);
  p[t]       = (x0 - mean) * inv * g[t] + bt[t];
  p[t + 256] = (x1 - mean) * inv * g[t + 256] + bt[t + 256];
}

// ---------------- host launch ---------------------------------------------------------
extern "C" void kernel_launch(void* const* d_in, const int* in_sizes, int n_in,
                              void* d_out, int out_size, void* d_ws, size_t ws_size,
                              hipStream_t stream) {
  (void)in_sizes; (void)n_in; (void)out_size; (void)ws_size;
  const float* query      = (const float*)d_in[0];
  const float* value      = (const float*)d_in[1];
  const float* prev_align = (const float*)d_in[2];
  const float* conv_w     = (const float*)d_in[3];
  const float* conv_b     = (const float*)d_in[4];
  const float* wq_w       = (const float*)d_in[5];
  const float* wq_b       = (const float*)d_in[6];
  const float* wv_w       = (const float*)d_in[7];
  const float* wv_b       = (const float*)d_in[8];
  const float* wu_w       = (const float*)d_in[9];
  const float* wu_b       = (const float*)d_in[10];
  const float* fc_w       = (const float*)d_in[11];
  const float* fc_b       = (const float*)d_in[12];
  const float* ln_g       = (const float*)d_in[13];
  const float* ln_b       = (const float*)d_in[14];

  float* out      = (float*)d_out;
  float* alignOut = out + (size_t)BB * SS * DD;     // tuple: (out, align)

  const size_t plane = (size_t)BB * SS * DD;        // 4.19M elements
  float*  ws   = (float*)d_ws;                      // ~70 MB total
  float*  loc  = ws;                                // plane f32
  __bf16* hb0  = (__bf16*)(ws + plane);             // bf16 arena
  __bf16* qb   = hb0;                               // plane halves each
  __bf16* vb   = hb0 + plane;
  __bf16* vT   = hb0 + 2 * plane;
  __bf16* qact = hb0 + 3 * plane;
  __bf16* vact = hb0 + 4 * plane;
  __bf16* ctxb = hb0 + 5 * plane;
  __bf16* wqT  = hb0 + 6 * plane;                   // 512x512
  __bf16* wvT  = wqT + (size_t)DD * DD;
  __bf16* fcT  = wvT + (size_t)DD * DD;             // 512 rows x 1024 k

  // Prep: one-shot bf16 conversions + weight transposes (outside hot loops)
  cvt_bf16_kernel<<<plane / 256, 256, 0, stream>>>(query, qact);
  cvt_bf16_kernel<<<plane / 256, 256, 0, stream>>>(value, vact);
  transpose_bf16_kernel<<<(DD * DD) / 256, 256, 0, stream>>>(wq_w, wqT, DD);
  transpose_bf16_kernel<<<(DD * DD) / 256, 256, 0, stream>>>(wv_w, wvT, DD);
  transpose_bf16_kernel<<<(2 * DD * DD) / 256, 256, 0, stream>>>(fc_w, fcT, 2 * DD);

  loc_kernel<<<(BB * HH * SS * DH) / 256, 256, 0, stream>>>(prev_align, conv_w, conv_b,
                                                            wu_w, wu_b, loc);
  proj_kernel<<<dim3(BB * SS / 128, DD / 64, 2), 256, 0, stream>>>(qact, vact, wqT, wvT,
                                                                   wq_b, wv_b, loc,
                                                                   qb, vb, vT);
  scores_kernel<<<dim3(SS / 128, SS / 64, HH * BB), 256, 0, stream>>>(qb, vb, alignOut);
  softmax_kernel<<<HH * BB * SS, 256, 0, stream>>>(alignOut);
  ctx_kernel<<<dim3(SS / 128, 1, HH * BB), 256, 0, stream>>>(alignOut, vT, ctxb);
  fc_kernel<<<dim3(BB * SS / 128, DD / 64), 256, 0, stream>>>(ctxb, qact, fcT, fc_b, out);
  ln_kernel<<<BB * SS, 256, 0, stream>>>(out, ln_g, ln_b);
}
